// NODE_58102317580472
// MI455X (gfx1250) — compile-verified, matrix operand order checked by reading, not run on hardware
//
#include <hip/hip_runtime.h>
#include <stdint.h>

// ---------------------------------------------------------------------------
// NODE scan for MI455X (gfx1250): persistent per-batch-tile workgroups,
// bf16 WMMA 16x16x32 with fp32 accumulation.
//  - W1/W3 per-wave slices pinned in VGPRs across all 1000 steps
//  - W2 streamed from L2 each step (opaque pointer blocks cross-step hoisting)
//  - A fragments software-pipelined one k-tile ahead out of LDS
// ---------------------------------------------------------------------------

typedef __attribute__((ext_vector_type(16))) __bf16 v16bf;
typedef __attribute__((ext_vector_type(8)))  __bf16 v8bf;
typedef __attribute__((ext_vector_type(8)))  float  v8f;

union BFrag { v16bf v; v8bf h[2]; };

enum { NB = 512, NT = 1000, NI = 16, NL = 128, NH = 512, NO = 64, KZ = 160 };

__device__ __forceinline__ v8f wmma_bf16(v16bf a, v16bf b, v8f c) {
  // (neg_a, A, neg_b, B, c_mod, C, reuse_a, reuse_b)
  return __builtin_amdgcn_wmma_f32_16x16x32_bf16(false, a, false, b,
                                                 (short)0, c, false, false);
}

// A fragment (16x32 bf16) from a row-major bf16 LDS buffer.
// ISA layout: lanes 0-15 hold M=0..15 / K-chunks {kt*32+0..7, +16..23};
//             lanes 16-31 hold the +8 K-chunks.
__device__ __forceinline__ v16bf load_a(const __bf16* buf, int stride, int kt,
                                        int lane) {
  const int m = lane & 15;
  const int base = kt * 32 + ((lane >> 4) << 3);
  BFrag f;
  f.h[0] = *(const v8bf*)(buf + m * stride + base);
  f.h[1] = *(const v8bf*)(buf + m * stride + base + 16);
  return f.v;
}

// B fragment from a pre-packed weight tile (1KB, lane-major, 16 bf16/lane).
__device__ __forceinline__ v16bf load_b(const __bf16* tile, int lane) {
  BFrag f;
  const v8bf* p = (const v8bf*)(tile + lane * 16);
  f.h[0] = p[0];
  f.h[1] = p[1];
  return f.v;
}

// Pack fp32 weight W[Kreal,N] into bf16 tiles of 512 elems (kt-major, then nt),
// each tile stored lane-major matching the WMMA B-operand register layout.
// Rows Kreal..Kpad are zero padding.
__global__ __launch_bounds__(256) void node_pack_w(const float* __restrict__ W,
                                                   __bf16* __restrict__ out,
                                                   int Kreal, int Kpad, int N) {
  int idx = blockIdx.x * 256 + threadIdx.x;
  int total = Kpad * N;
  if (idx >= total) return;
  int tile = idx >> 9;
  int lane = (idx >> 4) & 31;
  int e = idx & 15;
  int ntiles_n = N >> 4;
  int kt = tile / ntiles_n;
  int nt = tile - kt * ntiles_n;
  int koff = ((lane >> 4) << 3) + (e < 8 ? e : e + 8);
  int k = kt * 32 + koff;
  int n = nt * 16 + (lane & 15);
  float v = (k < Kreal) ? W[(size_t)k * N + n] : 0.0f;
  out[idx] = (__bf16)v;
}

// Persistent scan: one WG owns 16 batch rows for all T steps.
__global__ __launch_bounds__(256) void node_scan(
    const float* __restrict__ x,    // [B,T,I] fp32
    const __bf16* __restrict__ w1,  // packed [KZ x H]
    const float* __restrict__ b1,
    const __bf16* __restrict__ w2,  // packed [H x H]
    const float* __restrict__ b2,
    const __bf16* __restrict__ w3,  // packed [H x L]
    const float* __restrict__ b3,
    float* __restrict__ lat)        // [B,T,L] fp32
{
  __shared__ __bf16 zbuf[16 * KZ];   // [h(128) | x(16) | pad(16)] bf16
  __shared__ __bf16 abuf[16 * NH];   // post-GEMM1 activations
  __shared__ __bf16 a2buf[16 * NH];  // post-GEMM2 activations
  __shared__ float  hbuf[16 * NL];   // fp32 master state

  const int tid = threadIdx.x;
  const int wv = tid >> 5;      // wave 0..7
  const int lane = tid & 31;
  const int hi = lane >> 4;
  const int ln = lane & 15;
  const int b0 = blockIdx.x * 16;

  // Per-lane biases hoisted out of the T loop (columns per wave are fixed).
  float bias1[4], bias2[4];
  #pragma unroll
  for (int i = 0; i < 4; ++i) {
    bias1[i] = b1[(wv * 4 + i) * 16 + ln];
    bias2[i] = b2[(wv * 4 + i) * 16 + ln];
  }
  const float bias3 = b3[wv * 16 + ln];

  // Pin this wave's W1 slice (5 kt x 4 nt = 20 frags = 160 VGPRs) and
  // W3 slice (16 kt x 1 nt = 16 frags = 128 VGPRs) in registers for all steps.
  v16bf w1f[20];
  #pragma unroll
  for (int kt = 0; kt < 5; ++kt)
    #pragma unroll
    for (int i = 0; i < 4; ++i)
      w1f[kt * 4 + i] =
          load_b(w1 + ((size_t)(kt * (NH / 16) + wv * 4 + i) << 9), lane);
  v16bf w3f[16];
  #pragma unroll
  for (int kt = 0; kt < 16; ++kt)
    w3f[kt] = load_b(w3 + ((size_t)(kt * (NL / 16) + wv) << 9), lane);

  // Init state and load x_0.
  for (int i = tid; i < 16 * KZ; i += 256) zbuf[i] = (__bf16)0.0f;
  for (int i = tid; i < 16 * NL; i += 256) hbuf[i] = 0.0f;
  {
    int m = tid >> 4, c = tid & 15;  // 256 threads cover the 16x16 x-tile
    zbuf[m * KZ + NL + c] = (__bf16)x[((size_t)(b0 + m) * NT + 0) * NI + c];
  }

  uint64_t w2a = (uint64_t)w2;

  for (int t = 0; t < NT; ++t) {
    __syncthreads();  // z (h + x_t) ready

    // ---- GEMM1: [16,160] x [160,512] -> relu -> abuf (W1 in registers) ----
    {
      v8f acc[4] = {};
      v16bf a_cur = load_a(zbuf, KZ, 0, lane);
      #pragma unroll
      for (int kt = 0; kt < 5; ++kt) {
        v16bf a_nxt = a_cur;
        if (kt < 4) a_nxt = load_a(zbuf, KZ, kt + 1, lane);
        #pragma unroll
        for (int i = 0; i < 4; ++i)
          acc[i] = wmma_bf16(a_cur, w1f[kt * 4 + i], acc[i]);
        a_cur = a_nxt;
      }
      #pragma unroll
      for (int i = 0; i < 4; ++i)
        #pragma unroll
        for (int j = 0; j < 8; ++j) {
          float v = acc[i][j] + bias1[i];
          v = v > 0.0f ? v : 0.0f;
          abuf[(j + hi * 8) * NH + (wv * 4 + i) * 16 + ln] = (__bf16)v;
        }
    }
    __syncthreads();  // abuf ready

    // ---- GEMM2: [16,512] x [512,512] -> relu -> a2buf (W2 streamed) ----
    {
      // Opaque pointer: stops the compiler from hoisting W2 fragments into
      // VGPRs across the 1000-step loop (which caused scratch spills).
      asm volatile("" : "+s"(w2a));
      const __bf16* w2t = (const __bf16*)w2a;

      v8f acc[4] = {};
      v16bf a_cur = load_a(abuf, NH, 0, lane);
      #pragma unroll
      for (int kt = 0; kt < 16; ++kt) {
        v16bf a_nxt = a_cur;
        if (kt < 15) a_nxt = load_a(abuf, NH, kt + 1, lane);
        if (kt < 14)  // prefetch next-next k-tile group (1 byte/lane spans 1KB)
          __builtin_prefetch(
              w2t + ((size_t)((kt + 2) * (NH / 16) + wv * 4) << 9) + lane * 16,
              0, 1);
        #pragma unroll
        for (int i = 0; i < 4; ++i) {
          v16bf bf = load_b(
              w2t + ((size_t)(kt * (NH / 16) + wv * 4 + i) << 9), lane);
          acc[i] = wmma_bf16(a_cur, bf, acc[i]);
        }
        a_cur = a_nxt;
      }
      #pragma unroll
      for (int i = 0; i < 4; ++i)
        #pragma unroll
        for (int j = 0; j < 8; ++j) {
          float v = acc[i][j] + bias2[i];
          v = v > 0.0f ? v : 0.0f;
          a2buf[(j + hi * 8) * NH + (wv * 4 + i) * 16 + ln] = (__bf16)v;
        }
    }
    __syncthreads();  // a2buf ready; all GEMM1 reads of z done

    // Stage x_{t+1} into z (safe: z reads finished at previous barrier).
    if (t + 1 < NT) {
      int m = tid >> 4, c = tid & 15;
      zbuf[m * KZ + NL + c] =
          (__bf16)x[((size_t)(b0 + m) * NT + (t + 1)) * NI + c];
    }

    // ---- GEMM3: [16,512] x [512,128] (W3 in registers); h update ----
    {
      v8f acc = {};
      v16bf a_cur = load_a(a2buf, NH, 0, lane);
      #pragma unroll
      for (int kt = 0; kt < 16; ++kt) {
        v16bf a_nxt = a_cur;
        if (kt < 15) a_nxt = load_a(a2buf, NH, kt + 1, lane);
        acc = wmma_bf16(a_cur, w3f[kt], acc);
        a_cur = a_nxt;
      }
      #pragma unroll
      for (int j = 0; j < 8; ++j) {
        const int m = j + hi * 8;
        const int n = wv * 16 + ln;
        float hn = hbuf[m * NL + n] + 0.1f * (acc[j] + bias3);
        hbuf[m * NL + n] = hn;          // fp32 master
        zbuf[m * KZ + n] = (__bf16)hn;  // bf16 operand for next step
        lat[((size_t)(b0 + m) * NT + t) * NL + n] = hn;
      }
    }
  }
}

// output = latents[B*T,128] @ Wr[128,64] + br : embarrassingly parallel WMMA
__global__ __launch_bounds__(256) void node_proj(
    const float* __restrict__ lat,  // [B*T, L]
    const __bf16* __restrict__ wr,  // packed [L x O]
    const float* __restrict__ br,
    float* __restrict__ out)        // [B*T, O]
{
  const int wv = threadIdx.x >> 5;
  const int lane = threadIdx.x & 31;
  const int hi = lane >> 4;
  const int ln = lane & 15;
  const long mt = (long)blockIdx.x * 8 + wv;  // m-tile id
  const long r0 = mt * 16;

  // Build bf16 A fragments from fp32 latents.
  v16bf afr[4];
  #pragma unroll
  for (int kt = 0; kt < 4; ++kt) {
    const int base = kt * 32 + (hi << 3);
    const float* p = lat + (r0 + ln) * NL + base;
    BFrag f;
    #pragma unroll
    for (int e = 0; e < 8; ++e) {
      f.v[e] = (__bf16)p[e];
      f.v[e + 8] = (__bf16)p[e + 16];
    }
    afr[kt] = f.v;
  }

  #pragma unroll
  for (int nt = 0; nt < 4; ++nt) {
    v8f acc = {};
    #pragma unroll
    for (int kt = 0; kt < 4; ++kt) {
      v16bf bf = load_b(wr + (((size_t)kt * (NO / 16) + nt) << 9), lane);
      acc = wmma_bf16(afr[kt], bf, acc);
    }
    const float bias = br[nt * 16 + ln];
    #pragma unroll
    for (int j = 0; j < 8; ++j) {
      out[(r0 + j + hi * 8) * NO + nt * 16 + ln] = acc[j] + bias;
    }
  }
}

extern "C" void kernel_launch(void* const* d_in, const int* in_sizes, int n_in,
                              void* d_out, int out_size, void* d_ws,
                              size_t ws_size, hipStream_t stream) {
  const float* x  = (const float*)d_in[0];
  const float* W1 = (const float*)d_in[1];
  const float* b1 = (const float*)d_in[2];
  const float* W2 = (const float*)d_in[3];
  const float* b2 = (const float*)d_in[4];
  const float* W3 = (const float*)d_in[5];
  const float* b3 = (const float*)d_in[6];
  const float* Wr = (const float*)d_in[7];
  const float* br = (const float*)d_in[8];

  float* out = (float*)d_out;                        // [B,T,O]
  float* lat = out + (size_t)NB * NT * NO;           // [B,T,L]

  __bf16* w1pk = (__bf16*)d_ws;                      // 160*512
  __bf16* w2pk = w1pk + (size_t)KZ * NH;             // 512*512
  __bf16* w3pk = w2pk + (size_t)NH * NH;             // 512*128
  __bf16* wrpk = w3pk + (size_t)NH * NL;             // 128*64

  node_pack_w<<<(KZ * NH + 255) / 256, 256, 0, stream>>>(W1, w1pk, NL + NI, KZ, NH);
  node_pack_w<<<(NH * NH + 255) / 256, 256, 0, stream>>>(W2, w2pk, NH, NH, NH);
  node_pack_w<<<(NH * NL + 255) / 256, 256, 0, stream>>>(W3, w3pk, NH, NH, NL);
  node_pack_w<<<(NL * NO + 255) / 256, 256, 0, stream>>>(Wr, wrpk, NL, NL, NO);

  node_scan<<<NB / 16, 256, 0, stream>>>(x, w1pk, b1, w2pk, b2, w3pk, b3, lat);
  node_proj<<<(NB * NT / 16) / 8, 256, 0, stream>>>(lat, wrpk, br, out);
}
